// Block_19473381720396
// MI455X (gfx1250) — matrix operand order repaired and checked by reading, not compiled, not run on hardware
//
#include <hip/hip_runtime.h>
#include <hip/hip_bf16.h>
#include <math.h>

#define NVOX   200000
#define CH     96
#define KVOL   343
#define NEDGE  3200000

typedef __attribute__((ext_vector_type(16))) _Float16 v16h;
typedef __attribute__((ext_vector_type(8)))  _Float16 v8h;
typedef __attribute__((ext_vector_type(8)))  float    v8f;

// Branchless float-accurate GELU (exact-erf form, Abramowitz-Stegun 7.1.26,
// max abs err ~1.5e-7): one v_rcp_f32 + one v_exp_f32 + straight-line FMAs.
// No EXEC divergence -> keeps the WMMA issue stream clean.
__device__ __forceinline__ float gelu_exact(float x) {
    const float u  = 0.70710678118654752f * x;      // x / sqrt(2)
    const float ax = __builtin_fabsf(u);
    const float t  = __builtin_amdgcn_rcpf(__builtin_fmaf(0.3275911f, ax, 1.0f));
    float p = __builtin_fmaf(t, 1.061405429f, -1.453152027f);
    p = __builtin_fmaf(t, p, 1.421413741f);
    p = __builtin_fmaf(t, p, -0.284496736f);
    p = __builtin_fmaf(t, p, 0.254829592f);
    p = p * t;
    const float e    = __expf(-u * u);              // v_exp_f32
    float erfv = __builtin_fmaf(-p, e, 1.0f);       // erf(|u|)
    erfv = __builtin_copysignf(erfv, u);            // erf(u), branchless
    return 0.5f * x * (1.0f + erfv);
}

// ---------------------------------------------------------------------------
// Zero the conv accumulator buffer (float4-vectorized).
// total = NVOX*CH = 19,200,000 floats = 4,800,000 float4 = 18750 blocks x 256.
__global__ void __launch_bounds__(256) zero_x_kernel(float4* __restrict__ p) {
    const int i = blockIdx.x * 256 + threadIdx.x;
    p[i] = make_float4(0.f, 0.f, 0.f, 0.f);
}

// ---------------------------------------------------------------------------
// Pack w1 [96x384] and w2 [384x96] (fp32, row-major) into f16 WMMA B-fragment
// order:  element i (0..15) of lane L for (ktile, ntile) holds
//   k = ktile*32 + (L/16)*16 + i ,  n = ntile*16 + (L%16)
// w1: 24 ntiles x 3 ktiles ; w2: 6 ntiles x 12 ktiles. 36864 elems each.
__global__ void __launch_bounds__(256) pack_w_kernel(
        const float* __restrict__ w1, const float* __restrict__ w2,
        _Float16* __restrict__ w1p, _Float16* __restrict__ w2p) {
    const int g = blockIdx.x * 256 + threadIdx.x;     // 0 .. 73727
    if (g < 36864) {
        const int i    = g & 15;
        const int lane = (g >> 4) & 31;
        const int kt   = (g >> 9) % 3;
        const int nt   = g / 1536;
        const int k    = kt * 32 + (lane >> 4) * 16 + i;
        const int n    = nt * 16 + (lane & 15);
        w1p[g] = (_Float16)w1[k * 384 + n];
    } else {
        const int g2   = g - 36864;
        const int i    = g2 & 15;
        const int lane = (g2 >> 4) & 31;
        const int kt   = (g2 >> 9) % 12;
        const int nt   = g2 / 6144;
        const int k    = kt * 32 + (lane >> 4) * 16 + i;
        const int n    = nt * 16 + (lane & 15);
        w2p[g2] = (_Float16)w2[k * 96 + n];
    }
}

// ---------------------------------------------------------------------------
// Sparse depthwise conv: one wave per edge, 3 channels per lane.
// x[out,c] += feats[in,c] * w_dw[k,c]  via global_atomic_add_f32.
// feats (76.8 MB) and x (76.8 MB) are both L2-resident (192 MB L2).
__global__ void __launch_bounds__(256) edge_scatter_kernel(
        const float* __restrict__ feats, const float* __restrict__ w_dw,
        const int* __restrict__ in_idx, const int* __restrict__ out_idx,
        const int* __restrict__ k_idx, float* __restrict__ x) {
    const long long g = (long long)blockIdx.x * 256 + threadIdx.x;
    const int e    = (int)(g >> 5);
    const int lane = (int)(g & 31);
    const int vi = in_idx[e];
    const int vo = out_idx[e];
    const int kk = k_idx[e];
    const float* fi = feats + (size_t)vi * CH;
    const float* wk = w_dw + (size_t)kk * CH;
    float* xo = x + (size_t)vo * CH;
#pragma unroll
    for (int j = 0; j < 3; ++j) {
        const int c = lane + j * 32;
        unsafeAtomicAdd(xo + c, fi[c] * wk[c]);
    }
}

// ---------------------------------------------------------------------------
// Fused:  x += b_dw ; LayerNorm ; h = gelu(x@w1+b1) ; y = h@w2+b2 ; out = feats+y
// One wave32 per 16-row tile, 4 waves (128 thr) per block, 64 rows/block.
// N = 200000 = 3125 blocks * 64 rows exactly (no tail).
__global__ void __launch_bounds__(128) ln_mlp_kernel(
        const float* __restrict__ x, const float* __restrict__ feats,
        const float* __restrict__ b_dw,
        const float* __restrict__ ln_w, const float* __restrict__ ln_b,
        const float* __restrict__ b1, const float* __restrict__ b2,
        const _Float16* __restrict__ w1p, const _Float16* __restrict__ w2p,
        float* __restrict__ out) {
    __shared__ _Float16 sXn[4][16][96];    // 12 KB: normalized input tile (A of GEMM1)
    __shared__ _Float16 sH[4][16][384];    // 48 KB: hidden tile (A of GEMM2)

    const int tid  = threadIdx.x;
    const int wave = tid >> 5;
    const int lane = tid & 31;
    const int m    = lane & 15;            // row within tile
    const int hsel = lane >> 4;            // which half of K / channels
    const int rbase = (blockIdx.x * 4 + wave) * 16;

    // ---- bias + LayerNorm: 2 lanes per row, 48 channels each ----
    const float* xrow = x + (size_t)(rbase + m) * CH;
    const int c0 = hsel * 48;
    float sum = 0.f, ss = 0.f;
    float4 vbuf[12];
#pragma unroll
    for (int j = 0; j < 12; ++j) {
        float4 v        = *(const float4*)(xrow + c0 + j * 4);
        const float4 bb = *(const float4*)(b_dw + c0 + j * 4);
        v.x += bb.x; v.y += bb.y; v.z += bb.z; v.w += bb.w;
        vbuf[j] = v;
        sum += v.x + v.y + v.z + v.w;
        ss  += v.x * v.x + v.y * v.y + v.z * v.z + v.w * v.w;
    }
    sum += __shfl_xor(sum, 16, 32);
    ss  += __shfl_xor(ss, 16, 32);
    const float mu   = sum * (1.0f / 96.0f);
    const float rstd = rsqrtf(ss * (1.0f / 96.0f) - mu * mu + 1e-6f);
#pragma unroll
    for (int j = 0; j < 12; ++j) {
        const int c = c0 + j * 4;
        const float4 v = vbuf[j];
        sXn[wave][m][c + 0] = (_Float16)((v.x - mu) * rstd * ln_w[c + 0] + ln_b[c + 0]);
        sXn[wave][m][c + 1] = (_Float16)((v.y - mu) * rstd * ln_w[c + 1] + ln_b[c + 1]);
        sXn[wave][m][c + 2] = (_Float16)((v.z - mu) * rstd * ln_w[c + 2] + ln_b[c + 2]);
        sXn[wave][m][c + 3] = (_Float16)((v.w - mu) * rstd * ln_w[c + 3] + ln_b[c + 3]);
    }
    __syncthreads();

    // ---- GEMM1: [16x96] x [96x384], K in 3 steps of 32 ----
    v16h a1[3];
#pragma unroll
    for (int kt = 0; kt < 3; ++kt) {
        const int kb = kt * 32 + hsel * 16;
        const v8h lo = *(const v8h*)(&sXn[wave][m][kb]);
        const v8h hi = *(const v8h*)(&sXn[wave][m][kb + 8]);
#pragma unroll
        for (int i = 0; i < 8; ++i) { a1[kt][i] = lo[i]; a1[kt][8 + i] = hi[i]; }
    }
#pragma unroll
    for (int nt = 0; nt < 24; ++nt) {
        v8f acc = {};
#pragma unroll
        for (int kt = 0; kt < 3; ++kt) {
            const v16h b = *(const v16h*)(w1p + ((size_t)(nt * 3 + kt) * 32 + lane) * 16);
            acc = __builtin_amdgcn_wmma_f32_16x16x32_f16(
                false, a1[kt], false, b, (short)0, acc, false, false);
        }
        const int col   = nt * 16 + m;
        const float bia = b1[col];
#pragma unroll
        for (int v = 0; v < 8; ++v) {
            const float hv = gelu_exact(acc[v] + bia);
            sH[wave][v + hsel * 8][col] = (_Float16)hv;
        }
    }
    __syncthreads();

    // ---- GEMM2: [16x384] x [384x96], K in 12 steps, 6 n-tiles live ----
    v8f acc2[6];
#pragma unroll
    for (int nt = 0; nt < 6; ++nt) acc2[nt] = (v8f){};
#pragma unroll
    for (int kt = 0; kt < 12; ++kt) {
        const int kb = kt * 32 + hsel * 16;
        const v8h lo = *(const v8h*)(&sH[wave][m][kb]);
        const v8h hi = *(const v8h*)(&sH[wave][m][kb + 8]);
        v16h a;
#pragma unroll
        for (int i = 0; i < 8; ++i) { a[i] = lo[i]; a[8 + i] = hi[i]; }
#pragma unroll
        for (int nt = 0; nt < 6; ++nt) {
            const v16h b = *(const v16h*)(w2p + ((size_t)(nt * 12 + kt) * 32 + lane) * 16);
            acc2[nt] = __builtin_amdgcn_wmma_f32_16x16x32_f16(
                false, a, false, b, (short)0, acc2[nt], false, false);
        }
    }

    // ---- epilogue: bias + residual + store ----
#pragma unroll
    for (int nt = 0; nt < 6; ++nt) {
        const int col   = nt * 16 + m;
        const float bia = b2[col];
#pragma unroll
        for (int v = 0; v < 8; ++v) {
            const int row    = rbase + v + hsel * 8;
            const size_t off = (size_t)row * CH + col;
            out[off] = feats[off] + acc2[nt][v] + bia;
        }
    }
}

// ---------------------------------------------------------------------------
extern "C" void kernel_launch(void* const* d_in, const int* in_sizes, int n_in,
                              void* d_out, int out_size, void* d_ws, size_t ws_size,
                              hipStream_t stream) {
    const float* feats = (const float*)d_in[0];
    const float* w_dw  = (const float*)d_in[1];
    const float* b_dw  = (const float*)d_in[2];
    const float* ln_w  = (const float*)d_in[3];
    const float* ln_b  = (const float*)d_in[4];
    const float* w1    = (const float*)d_in[5];
    const float* b1    = (const float*)d_in[6];
    const float* w2    = (const float*)d_in[7];
    const float* b2    = (const float*)d_in[8];
    const int* in_idx  = (const int*)d_in[9];
    const int* out_idx = (const int*)d_in[10];
    const int* k_idx   = (const int*)d_in[11];
    float* out = (float*)d_out;

    // workspace layout
    float* d_x = (float*)d_ws;                                   // NVOX*CH f32 = 76.8 MB
    const size_t xbytes = (size_t)NVOX * CH * sizeof(float);     // 16B aligned
    _Float16* w1p = (_Float16*)((char*)d_ws + xbytes);           // 36864 f16
    _Float16* w2p = w1p + 36864;                                 // 36864 f16

    // 1) zero conv accumulator: 4.8M float4
    zero_x_kernel<<<dim3(18750), dim3(256), 0, stream>>>((float4*)d_x);
    // 2) pack weights into WMMA B-fragment order (f16)
    pack_w_kernel<<<dim3(288), dim3(256), 0, stream>>>(w1, w2, w1p, w2p);
    // 3) sparse depthwise conv scatter: one wave per edge
    edge_scatter_kernel<<<dim3(400000), dim3(256), 0, stream>>>(
        feats, w_dw, in_idx, out_idx, k_idx, d_x);
    // 4) fused bias + LayerNorm + MLP (WMMA) + residual
    ln_mlp_kernel<<<dim3(3125), dim3(128), 0, stream>>>(
        d_x, feats, b_dw, ln_w, ln_b, b1, b2, w1p, w2p, out);
}
